// DeepGCN_Dyn_12841952215496
// MI455X (gfx1250) — compile-verified, matrix operand order checked by reading, not run on hardware
//
#include <hip/hip_runtime.h>
#include <hip/hip_bf16.h>

#define DEV_B   4
#define DEV_N   4096
#define DEV_NF  16
#define DEV_K   20
#define BN_EPS_F 1e-5f

typedef __attribute__((ext_vector_type(16))) _Float16 v16h;
typedef __attribute__((ext_vector_type(8)))  float    v8f;
typedef __attribute__((ext_vector_type(4)))  float    v4f;

// Wave-local LDS sync: same-wave LDS ops are in-order in hardware; the
// barrier + clobber stops the compiler caching LDS values across iterations.
__device__ inline void dgcn_wave_sync() {
  __builtin_amdgcn_wave_barrier();
  asm volatile("" ::: "memory");
}

// ---------------------------------------------------------------------------
// Prepare head round: inputs [B,4,N,1] -> xf[B,N,4] (f32 for edgeconv),
// fh[B,N,16] (f16, ch0..2 = xyz, rest 0, for WMMA knn), x2[B,N] = |xyz|^2.
// ---------------------------------------------------------------------------
__global__ void dgcn_prepare_head(const float* __restrict__ inp,
                                  float* __restrict__ xf,
                                  _Float16* __restrict__ fh,
                                  float* __restrict__ x2) {
  int i = blockIdx.x * blockDim.x + threadIdx.x;
  if (i >= DEV_B * DEV_N) return;
  int b = i / DEV_N, n = i % DEV_N;
  const float* p = inp + (size_t)b * 4 * DEV_N + n;
  float c0 = p[0 * DEV_N], c1 = p[1 * DEV_N], c2 = p[2 * DEV_N], c3 = p[3 * DEV_N];
  float* xo = xf + (size_t)i * 4;
  xo[0] = c0; xo[1] = c1; xo[2] = c2; xo[3] = c3;
  _Float16* fo = fh + (size_t)i * DEV_NF;
#pragma unroll
  for (int j = 0; j < DEV_NF; ++j) fo[j] = (_Float16)0.f;
  fo[0] = (_Float16)c0; fo[1] = (_Float16)c1; fo[2] = (_Float16)c2;
  x2[i] = c0 * c0 + c1 * c1 + c2 * c2;
}

// ---------------------------------------------------------------------------
// Convert current f32 features -> f16 + squared norms (for block-round knn).
// ---------------------------------------------------------------------------
__global__ void dgcn_convert(const float* __restrict__ feat,
                             _Float16* __restrict__ fh,
                             float* __restrict__ x2) {
  int i = blockIdx.x * blockDim.x + threadIdx.x;
  if (i >= DEV_B * DEV_N) return;
  const float* fi = feat + (size_t)i * DEV_NF;
  _Float16* fo = fh + (size_t)i * DEV_NF;
  float s = 0.f;
#pragma unroll
  for (int j = 0; j < DEV_NF; ++j) {
    float v = fi[j];
    fo[j] = (_Float16)v;
    s += v * v;
  }
  x2[i] = s;
}

// ---------------------------------------------------------------------------
// Fused KNN: WMMA Gram tiles (16x16, K=16 padded to 32) -> dist slab in LDS
// -> per-row iterative min-extraction (sorted ascending, every dstep-th kept).
// Grid: (N/16, B). Block: 256 (8 waves). Dynamic LDS: 16*N*4 + 512 bytes.
// ---------------------------------------------------------------------------
__global__ void __launch_bounds__(256)
dgcn_knn(const _Float16* __restrict__ fh,
         const float* __restrict__ x2g,
         int* __restrict__ topo,
         int kd, int dstep) {
  extern __shared__ char dgcn_smem[];
  float*    dist  = (float*)dgcn_smem;                       // 16 * 4096 f32
  _Float16* afeat = (_Float16*)(dgcn_smem + 16 * DEV_N * 4); // 16 rows * 16 f16

  const int t = threadIdx.x, lane = t & 31, w = t >> 5;
  const int batch = blockIdx.y, rowbase = blockIdx.x * 16;
  const int nloc = lane & 15, hi = lane >> 4;

  afeat[t] = fh[((size_t)batch * DEV_N + rowbase + (t >> 4)) * DEV_NF + (t & 15)];
  __syncthreads();

  // Row norms for this lane's 8 output rows -> registers (two b128 loads).
  float rx2[8];
  {
    const float* rp = x2g + batch * DEV_N + rowbase + (hi ? 8 : 0);
#pragma unroll
    for (int r = 0; r < 8; ++r) rx2[r] = rp[r];
  }

  // A fragment (16x32 f16, K 16..31 zero). Lanes 0-15: K0-7, lanes 16-31: K8-15.
  v16h a = {};
  {
    const _Float16* ap = afeat + nloc * DEV_NF + (hi ? 8 : 0);
#pragma unroll
    for (int h = 0; h < 8; ++h) a[h] = ap[h];
  }

  // Each wave handles 32 contiguous column tiles, unrolled x2.
  const _Float16* fbase = fh + (size_t)batch * DEV_N * DEV_NF;
  const float* x2base = x2g + batch * DEV_N;
  const int ct0 = w * 32;
  for (int tt = 0; tt < 32; tt += 2) {
    int cb0 = (ct0 + tt) * 16;
    int cb1 = cb0 + 16;
    int pcol = (cb0 + 64 < DEV_N) ? cb0 + 64 : cb0;  // branchless clamp
    __builtin_prefetch(fbase + (size_t)pcol * DEV_NF, 0, 1);

    v16h b0 = {}, b1 = {};
    const _Float16* bp0 = fbase + (size_t)(cb0 + nloc) * DEV_NF + (hi ? 8 : 0);
    const _Float16* bp1 = fbase + (size_t)(cb1 + nloc) * DEV_NF + (hi ? 8 : 0);
#pragma unroll
    for (int h = 0; h < 8; ++h) b0[h] = bp0[h];
#pragma unroll
    for (int h = 0; h < 8; ++h) b1[h] = bp1[h];

    v8f c0 = {}, c1 = {};
    c0 = __builtin_amdgcn_wmma_f32_16x16x32_f16(false, a, false, b0,
                                                (short)0, c0, false, false);
    c1 = __builtin_amdgcn_wmma_f32_16x16x32_f16(false, a, false, b1,
                                                (short)0, c1, false, false);

    float x2c0 = x2base[cb0 + nloc];
    float x2c1 = x2base[cb1 + nloc];
#pragma unroll
    for (int r = 0; r < 8; ++r) {
      int m = r + (hi ? 8 : 0);
      dist[m * DEV_N + cb0 + nloc] = rx2[r] + x2c0 - 2.0f * c0[r];
      dist[m * DEV_N + cb1 + nloc] = rx2[r] + x2c1 - 2.0f * c1[r];
    }
  }
  __syncthreads();

  // Selection: wave w owns rows w and w+8; purely wave-local.
  // Each lane scans a contiguous 128-float chunk with b128 LDS loads.
  for (int rr = 0; rr < 2; ++rr) {
    int row = w + rr * 8;
    float* rd = dist + row * DEV_N;
    const v4f* rd4 = (const v4f*)(rd + lane * 128);
    int* orow = topo + ((size_t)batch * DEV_N + rowbase + row) * DEV_K;
    int jb0 = lane * 128;
    for (int sel = 0; sel < kd; ++sel) {
      float bv = 3.0e38f;
      int bi = 0x7fffffff;
#pragma unroll 4
      for (int it = 0; it < 32; ++it) {
        v4f d = rd4[it];
        int jb = jb0 + it * 4;
        if (d.x < bv) { bv = d.x; bi = jb; }
        if (d.y < bv) { bv = d.y; bi = jb + 1; }
        if (d.z < bv) { bv = d.z; bi = jb + 2; }
        if (d.w < bv) { bv = d.w; bi = jb + 3; }
      }
#pragma unroll
      for (int off = 16; off; off >>= 1) {
        float ov = __shfl_xor(bv, off, 32);
        int   oi = __shfl_xor(bi, off, 32);
        if (ov < bv || (ov == bv && oi < bi)) { bv = ov; bi = oi; }
      }
      if (lane == 0) {
        rd[bi] = 3.0e38f;
        if ((sel % dstep) == 0) orow[sel / dstep] = bi;
      }
      dgcn_wave_sync();
    }
  }
}

// ---------------------------------------------------------------------------
// EdgeConv head: C=4, W[16,8]. out[b,n,o] = max_k relu(bn(W.[xi, xj-xi]+b)).
// ---------------------------------------------------------------------------
__global__ void dgcn_econv_head(const float* __restrict__ xf,
                                const int* __restrict__ topo,
                                const float* __restrict__ W,
                                const float* __restrict__ bias,
                                const float* __restrict__ g,
                                const float* __restrict__ be,
                                float* __restrict__ out) {
  int t = blockIdx.x * blockDim.x + threadIdx.x;
  if (t >= DEV_B * DEV_N * DEV_NF) return;
  int o = t & 15, n = (t >> 4) & (DEV_N - 1), b = t >> 16;
  const float* xi = xf + ((size_t)b * DEV_N + n) * 4;
  const int* nb = topo + ((size_t)b * DEV_N + n) * DEV_K;
  const float* w = W + o * 8;
  float scale = g[o] * rsqrtf(1.f + BN_EPS_F);
  float beo = be[o];
  float xi0 = xi[0], xi1 = xi[1], xi2 = xi[2], xi3 = xi[3];
  float base = bias[o] + w[0] * xi0 + w[1] * xi1 + w[2] * xi2 + w[3] * xi3;
  float best = 0.f;
  for (int j = 0; j < DEV_K; ++j) {
    const float* xj = xf + ((size_t)b * DEV_N + nb[j]) * 4;
    float h = base + w[4] * (xj[0] - xi0) + w[5] * (xj[1] - xi1) +
              w[6] * (xj[2] - xi2) + w[7] * (xj[3] - xi3);
    h = scale * h + beo;
    best = fmaxf(best, fmaxf(h, 0.f));
  }
  out[t] = best;
}

// ---------------------------------------------------------------------------
// EdgeConv block: C=16, W[16,32], with residual add.
// ---------------------------------------------------------------------------
__global__ void dgcn_econv_block(const float* __restrict__ fin,
                                 const int* __restrict__ topo,
                                 const float* __restrict__ W,
                                 const float* __restrict__ bias,
                                 const float* __restrict__ g,
                                 const float* __restrict__ be,
                                 float* __restrict__ out) {
  int t = blockIdx.x * blockDim.x + threadIdx.x;
  if (t >= DEV_B * DEV_N * DEV_NF) return;
  int o = t & 15, n = (t >> 4) & (DEV_N - 1), b = t >> 16;
  const float* fi = fin + ((size_t)b * DEV_N + n) * DEV_NF;
  const int* nb = topo + ((size_t)b * DEV_N + n) * DEV_K;
  const float* w = W + o * 32;
  float fival[DEV_NF];
  float base = bias[o];
#pragma unroll
  for (int c = 0; c < DEV_NF; ++c) {
    fival[c] = fi[c];
    base += w[c] * fival[c];
  }
  float scale = g[o] * rsqrtf(1.f + BN_EPS_F);
  float beo = be[o];
  float best = 0.f;
  for (int j = 0; j < DEV_K; ++j) {
    const float* fj = fin + ((size_t)b * DEV_N + nb[j]) * DEV_NF;
    float h = base;
#pragma unroll
    for (int c = 0; c < DEV_NF; ++c) h += w[16 + c] * (fj[c] - fival[c]);
    h = scale * h + beo;
    best = fmaxf(best, fmaxf(h, 0.f));
  }
  out[t] = fival[o] + best;
}

// ---------------------------------------------------------------------------
// Host launcher.
// ---------------------------------------------------------------------------
extern "C" void kernel_launch(void* const* d_in, const int* in_sizes, int n_in,
                              void* d_out, int out_size, void* d_ws, size_t ws_size,
                              hipStream_t stream) {
  (void)in_sizes; (void)n_in; (void)out_size; (void)ws_size;

  const float* inp     = (const float*)d_in[0];
  const float* head_w  = (const float*)d_in[1];
  const float* head_b  = (const float*)d_in[2];
  const float* head_g  = (const float*)d_in[3];
  const float* head_be = (const float*)d_in[4];
  const float* bb_w    = (const float*)d_in[5];
  const float* bb_b    = (const float*)d_in[6];
  const float* bb_g    = (const float*)d_in[7];
  const float* bb_be   = (const float*)d_in[8];

  float* out = (float*)d_out;
  const int BN    = DEV_B * DEV_N;       // 16384
  const int FEAT  = BN * DEV_NF;         // 262144
  const int TOPO1 = BN * DEV_K;          // 327680 ints per round

  float* ws    = (float*)d_ws;
  float* featA = ws;                          // FEAT f32
  float* featB = ws + FEAT;                   // FEAT f32
  float* xf    = ws + 2 * FEAT;               // BN*4 f32
  float* x2    = ws + 2 * FEAT + BN * 4;      // BN f32
  _Float16* fh = (_Float16*)(ws + 2 * FEAT + BN * 4 + BN); // BN*16 f16

  int* topo = (int*)(out + FEAT);             // [7][B][N][K] int32

  const int knn_smem = 16 * DEV_N * 4 + 512;  // 262656 B dynamic LDS
  dim3 b256(256);
  dim3 gBN((BN + 255) / 256);
  dim3 gFeat((FEAT + 255) / 256);
  dim3 gKnn(DEV_N / 16, DEV_B);

  // Head round (knn on xyz, d=1)
  dgcn_prepare_head<<<gBN, b256, 0, stream>>>(inp, xf, fh, x2);
  dgcn_knn<<<gKnn, b256, knn_smem, stream>>>(fh, x2, topo, DEV_K, 1);
  dgcn_econv_head<<<gFeat, b256, 0, stream>>>(xf, topo, head_w, head_b,
                                              head_g, head_be, featA);

  // Backbone rounds: d = 1+i, kd = 20*(1+i)
  float* cur = featA;
  float* nxt = featB;
  for (int i = 0; i < 6; ++i) {
    dgcn_convert<<<gBN, b256, 0, stream>>>(cur, fh, x2);
    dgcn_knn<<<gKnn, b256, knn_smem, stream>>>(
        fh, x2, topo + (size_t)(i + 1) * TOPO1, DEV_K * (i + 1), i + 1);
    float* dst = (i == 5) ? out : nxt;
    dgcn_econv_block<<<gFeat, b256, 0, stream>>>(
        cur, topo + (size_t)(i + 1) * TOPO1,
        bb_w + (size_t)i * DEV_NF * 2 * DEV_NF,
        bb_b + i * DEV_NF, bb_g + i * DEV_NF, bb_be + i * DEV_NF, dst);
    float* old = cur; cur = dst; nxt = old;
  }
}
